// OptimizedSegmentedAttention_704374636631
// MI455X (gfx1250) — matrix operand order repaired and checked by reading, not compiled, not run on hardware
//
#include <hip/hip_runtime.h>
#include <hip/hip_bf16.h>
#include <math.h>

// ---------------------------------------------------------------------------
// Problem constants (B=1)
// ---------------------------------------------------------------------------
#define SQ    4096      // sequence length
#define DIMM  512       // model dim
#define HEADS 8
#define DH    64
#define SEGL  512       // segment length
#define NPM   16        // persistent-memory tokens
#define KLTOT (NPM + SQ)     // 4112 rows in k/v (pm + seq)
#define KPAD  576            // padded keys per segment (16 pm + 512 + 48 pad), 18*32
#define CHUNK 192            // keys per LDS chunk (6 * 32)
#define NCHUNK 3

typedef __attribute__((ext_vector_type(16))) _Float16 v16h;
typedef __attribute__((ext_vector_type(8)))  _Float16 v8h;
typedef __attribute__((ext_vector_type(8)))  float    v8f;

union FragA { v16h v; v8h h[2]; };

__device__ __forceinline__ v8f wmma_f16(v16h a, v16h b, v8f c) {
  // D = A(16x32 f16) * B(32x16 f16) + C(16x16 f32)
  return __builtin_amdgcn_wmma_f32_16x16x32_f16(
      /*neg_a=*/false, a, /*neg_b=*/false, b,
      /*c_mod=*/(short)0, c, /*reuse_a=*/false, /*reuse_b=*/false);
}

// ---------------------------------------------------------------------------
// 1) RMSNorm: x = seq * rsqrt(mean(seq^2)+eps) * norm_w  ->  f16
//    one block per row (512 dims, 256 threads, 2 elems/thread)
// ---------------------------------------------------------------------------
__global__ __launch_bounds__(256)
void rmsnorm_kernel(const float* __restrict__ seq, const float* __restrict__ nw,
                    _Float16* __restrict__ xh) {
  int row = blockIdx.x;
  int t   = threadIdx.x;
  const float* p = seq + (size_t)row * DIMM;
  float v0 = p[2 * t], v1 = p[2 * t + 1];
  float ss = v0 * v0 + v1 * v1;
#pragma unroll
  for (int m = 16; m >= 1; m >>= 1) ss += __shfl_xor(ss, m, 32);
  __shared__ float red[8];
  int wid = t >> 5, lane = t & 31;
  if (lane == 0) red[wid] = ss;
  __syncthreads();
  float tot = red[0] + red[1] + red[2] + red[3] + red[4] + red[5] + red[6] + red[7];
  float sc = rsqrtf(tot * (1.0f / DIMM) + 1.1920929e-07f);
  xh[(size_t)row * DIMM + 2 * t]     = (_Float16)(v0 * sc * nw[2 * t]);
  xh[(size_t)row * DIMM + 2 * t + 1] = (_Float16)(v1 * sc * nw[2 * t + 1]);
}

// ---------------------------------------------------------------------------
// 2) Transpose + f32->f16 for weights: src (K x N) row-major -> dst (N x K)
// ---------------------------------------------------------------------------
__global__ __launch_bounds__(256)
void cvt_transpose_kernel(const float* __restrict__ src, _Float16* __restrict__ dst,
                          int K, int N) {
  int e = blockIdx.x * blockDim.x + threadIdx.x;
  if (e >= K * N) return;
  int n = e / K, k = e - n * K;
  dst[(size_t)n * K + k] = (_Float16)src[(size_t)k * N + n];
}

// ---------------------------------------------------------------------------
// 3) Persistent-memory tokens: rope K part at positions 0..15, copy V part.
//    pm shape (2, HEADS, NPM, DH). Writes rows [0,16) of k_h / v_h per head.
// ---------------------------------------------------------------------------
__global__ __launch_bounds__(256)
void prep_pm_kernel(const float* __restrict__ pm,
                    _Float16* __restrict__ kh, _Float16* __restrict__ vh) {
  int e = blockIdx.x * blockDim.x + threadIdx.x;  // HEADS*NPM*32 = 4096
  int h   = e >> 9;
  int row = (e >> 5) & 15;
  int d2  = e & 31;
  const float* pk = pm + (((size_t)0 * HEADS + h) * NPM + row) * DH;
  const float* pv = pm + (((size_t)1 * HEADS + h) * NPM + row) * DH;
  float x1 = pk[2 * d2], x2 = pk[2 * d2 + 1];
  float invf = powf(10000.0f, -((float)(2 * d2) / (float)DH));
  float ang = (float)row * invf;    // key positions 0..15 for pm tokens
  float sn, cs;
  sincosf(ang, &sn, &cs);
  size_t o = (((size_t)h * KLTOT) + row) * DH + 2 * d2;
  kh[o]     = (_Float16)(x1 * cs - x2 * sn);
  kh[o + 1] = (_Float16)(x2 * cs + x1 * sn);
  vh[o]     = (_Float16)pv[2 * d2];
  vh[o + 1] = (_Float16)pv[2 * d2 + 1];
}

// ---------------------------------------------------------------------------
// 4) RoPE + head split of fused qkv (S x 1536 f16):
//    q_h (H, S, DH) roped at pos 16+i ; k_h rows 16+i roped at pos 16+i ; v copy
// ---------------------------------------------------------------------------
__global__ __launch_bounds__(256)
void rope_split_kernel(const _Float16* __restrict__ qkv,
                       _Float16* __restrict__ qh, _Float16* __restrict__ kh,
                       _Float16* __restrict__ vh) {
  int i = blockIdx.x;            // token
  int e = threadIdx.x;           // 256 = HEADS * 32 pairs
  int h = e >> 5, d2 = e & 31;
  const _Float16* base = qkv + (size_t)i * (3 * DIMM);
  float q1 = (float)base[h * DH + 2 * d2];
  float q2 = (float)base[h * DH + 2 * d2 + 1];
  float k1 = (float)base[DIMM + h * DH + 2 * d2];
  float k2 = (float)base[DIMM + h * DH + 2 * d2 + 1];
  float invf = powf(10000.0f, -((float)(2 * d2) / (float)DH));
  float ang = (float)(NPM + i) * invf;    // q and key-row (16+i) share position
  float sn, cs;
  sincosf(ang, &sn, &cs);
  size_t qo = (((size_t)h * SQ) + i) * DH + 2 * d2;
  qh[qo]     = (_Float16)(q1 * cs - q2 * sn);
  qh[qo + 1] = (_Float16)(q2 * cs + q1 * sn);
  size_t ko = (((size_t)h * KLTOT) + NPM + i) * DH + 2 * d2;
  kh[ko]     = (_Float16)(k1 * cs - k2 * sn);
  kh[ko + 1] = (_Float16)(k2 * cs + k1 * sn);
  vh[ko]     = base[2 * DIMM + h * DH + 2 * d2];
  vh[ko + 1] = base[2 * DIMM + h * DH + 2 * d2 + 1];
}

// ---------------------------------------------------------------------------
// 5) WMMA GEMM: C(MxN) = A(MxK, f16 row-major) * Bt(NxK, f16 row-major)^T
//    Block tile 128x128, 8 waves, wave tile 64x32 (4x2 wmma tiles), K-step 32.
// ---------------------------------------------------------------------------
template <bool OUT_F32>
__global__ __launch_bounds__(256)
void gemm_wmma_kernel(const _Float16* __restrict__ A, const _Float16* __restrict__ Bt,
                      void* __restrict__ Cout, int M, int N, int K) {
  __shared__ _Float16 As[128 * 32];
  __shared__ _Float16 Bs[128 * 32];
  int tid = threadIdx.x;
  int w = tid >> 5, lane = tid & 31;
  int lanen = lane & 15, g = lane >> 4;
  int mblock = blockIdx.y * 128;
  int nblock = blockIdx.x * 128;
  int wm = (w >> 2) * 64;
  int wn = (w & 3) * 32;

  v8f acc[4][2];
#pragma unroll
  for (int a = 0; a < 4; a++)
#pragma unroll
    for (int b = 0; b < 2; b++)
#pragma unroll
      for (int r = 0; r < 8; r++) acc[a][b][r] = 0.0f;

  for (int kb = 0; kb < K; kb += 32) {
    __syncthreads();
#pragma unroll
    for (int i = 0; i < 2; i++) {
      int c = tid + i * 256;          // 512 b128 chunks per tile
      int row = c >> 2, c8 = (c & 3) * 8;
      *(v8h*)&As[row * 32 + c8] = *(const v8h*)&A[(size_t)(mblock + row) * K + kb + c8];
      *(v8h*)&Bs[row * 32 + c8] = *(const v8h*)&Bt[(size_t)(nblock + row) * K + kb + c8];
    }
    if (kb + 32 < K) {
      __builtin_prefetch(&A[(size_t)(mblock + (tid >> 1)) * K + kb + 32], 0, 1);
      __builtin_prefetch(&Bt[(size_t)(nblock + (tid >> 1)) * K + kb + 32], 0, 1);
    }
    __syncthreads();

    FragA af[4], bf[2];
#pragma unroll
    for (int tm = 0; tm < 4; tm++) {
      const _Float16* ap = &As[(wm + tm * 16 + lanen) * 32];
      af[tm].h[0] = *(const v8h*)&ap[g * 8];         // K = g*8 .. g*8+7
      af[tm].h[1] = *(const v8h*)&ap[16 + g * 8];    // K = 16+g*8 ..
    }
#pragma unroll
    for (int tn = 0; tn < 2; tn++) {
      const _Float16* bp = &Bs[(wn + tn * 16 + lanen) * 32];
      bf[tn].h[0] = *(const v8h*)&bp[g * 8];         // K = g*8 .. g*8+7
      bf[tn].h[1] = *(const v8h*)&bp[16 + g * 8];    // K = 16+g*8 ..
    }
#pragma unroll
    for (int tm = 0; tm < 4; tm++)
#pragma unroll
      for (int tn = 0; tn < 2; tn++)
        acc[tm][tn] = wmma_f16(af[tm].v, bf[tn].v, acc[tm][tn]);
  }

#pragma unroll
  for (int tm = 0; tm < 4; tm++)
#pragma unroll
    for (int tn = 0; tn < 2; tn++) {
      int col = nblock + wn + tn * 16 + lanen;
#pragma unroll
      for (int r = 0; r < 8; r++) {
        int row = mblock + wm + tm * 16 + r + 8 * g;
        if (OUT_F32)
          ((float*)Cout)[(size_t)row * N + col] = acc[tm][tn][r];
        else
          ((_Float16*)Cout)[(size_t)row * N + col] = (_Float16)acc[tm][tn][r];
      }
    }
}

// ---------------------------------------------------------------------------
// 6) Segment-local flash attention.
//    grid = (4 qblocks, 8 segs, 8 heads), block = 256 (8 waves).
//    Each wave owns 16 queries; keys streamed through LDS in 3 chunks of 192.
// ---------------------------------------------------------------------------
__global__ __launch_bounds__(256)
void attn_kernel(const _Float16* __restrict__ qh, const _Float16* __restrict__ kh,
                 const _Float16* __restrict__ vh, _Float16* __restrict__ ao) {
  __shared__ _Float16 Ks[CHUNK * DH];   // keys row-major (key x feature)
  __shared__ _Float16 Vt[DH * CHUNK];   // V transposed  (feature x key)
  __shared__ _Float16 Ps[8 * 16 * 32];  // per-wave P staging (16 x 32)

  int tid = threadIdx.x;
  int w = tid >> 5, lane = tid & 31;
  int lanen = lane & 15, g = lane >> 4;
  int qblk = blockIdx.x;   // 0..3
  int seg  = blockIdx.y;   // 0..7
  int head = blockIdx.z;   // 0..7
  int qb = qblk * 128 + w * 16;   // in-segment query base for this wave

  // Q fragments for DH=64 (two K=32 slabs), A-matrix layout
  FragA qf[2];
  {
    const _Float16* qrow = qh + ((size_t)head * SQ + seg * SEGL + qb + lanen) * DH;
    qf[0].h[0] = *(const v8h*)&qrow[g * 8];
    qf[0].h[1] = *(const v8h*)&qrow[16 + g * 8];
    qf[1].h[0] = *(const v8h*)&qrow[32 + g * 8];
    qf[1].h[1] = *(const v8h*)&qrow[48 + g * 8];
  }

  v8f acc[4];
  float mst[8], lst[8];
#pragma unroll
  for (int nb = 0; nb < 4; nb++)
#pragma unroll
    for (int r = 0; r < 8; r++) acc[nb][r] = 0.0f;
#pragma unroll
  for (int r = 0; r < 8; r++) { mst[r] = -3.0e38f; lst[r] = 0.0f; }

  const float NEG = -3.0e38f;

  for (int ch = 0; ch < NCHUNK; ch++) {
    int KB = ch * CHUNK;
    __syncthreads();   // previous chunk's compute done before overwrite
    // Load K chunk (row-major) and V chunk (transposed) into LDS; pad with 0.
#pragma unroll
    for (int i = 0; i < 6; i++) {
      int c = tid + i * 256;          // 1536 b128 chunks
      int row = c >> 3;               // 0..191 (key within chunk)
      int c8  = (c & 7) * 8;          // feature offset
      int kidx = KB + row;            // combined key index (pm|seq|pad)
      v8h kv, vv;
      if (kidx < NPM + SEGL) {
        size_t sr = (kidx < NPM) ? (size_t)kidx
                                 : (size_t)(NPM + seg * SEGL + (kidx - NPM));
        kv = *(const v8h*)&kh[((size_t)head * KLTOT + sr) * DH + c8];
        vv = *(const v8h*)&vh[((size_t)head * KLTOT + sr) * DH + c8];
      } else {
#pragma unroll
        for (int j = 0; j < 8; j++) { kv[j] = (_Float16)0.0f; vv[j] = (_Float16)0.0f; }
      }
      *(v8h*)&Ks[row * DH + c8] = kv;
#pragma unroll
      for (int j = 0; j < 8; j++) Vt[(c8 + j) * CHUNK + row] = vv[j];
    }
    __syncthreads();

    for (int kt = 0; kt < CHUNK; kt += 32) {
      // ---- scores: two 16-key tiles, each QK^T over DH=64 (2 wmma) ----
      v8f s[2];
#pragma unroll
      for (int t = 0; t < 2; t++) {
        const _Float16* kp = &Ks[(kt + t * 16 + lanen) * DH];
        FragA b0, b1;
        b0.h[0] = *(const v8h*)&kp[g * 8];           // features 0..31
        b0.h[1] = *(const v8h*)&kp[16 + g * 8];
        b1.h[0] = *(const v8h*)&kp[32 + g * 8];      // features 32..63
        b1.h[1] = *(const v8h*)&kp[48 + g * 8];
        v8f z;
#pragma unroll
        for (int r = 0; r < 8; r++) z[r] = 0.0f;
        z = wmma_f16(qf[0].v, b0.v, z);
        z = wmma_f16(qf[1].v, b1.v, z);
        s[t] = z;
      }
      // ---- mask (pm always visible; seq keys: causal within segment) ----
      int kidx0 = KB + kt + lanen;
      int kidx1 = kidx0 + 16;
#pragma unroll
      for (int r = 0; r < 8; r++) {
        int qrel = qb + r + 8 * g;
        bool v0 = (kidx0 < NPM) || ((kidx0 - NPM) <= qrel);  // pad keys auto-fail
        bool v1 = (kidx1 < NPM) || ((kidx1 - NPM) <= qrel);
        s[0][r] = v0 ? s[0][r] * 0.125f : NEG;   // 1/sqrt(64)
        s[1][r] = v1 ? s[1][r] * 0.125f : NEG;
      }
      // ---- online softmax update + stage P into LDS (f16) ----
#pragma unroll
      for (int r = 0; r < 8; r++) {
        float mx = fmaxf(s[0][r], s[1][r]);
#pragma unroll
        for (int m = 8; m >= 1; m >>= 1) mx = fmaxf(mx, __shfl_xor(mx, m, 32));
        float nm = fmaxf(mst[r], mx);
        float p0 = __expf(s[0][r] - nm);
        float p1 = __expf(s[1][r] - nm);
        float rs = p0 + p1;
#pragma unroll
        for (int m = 8; m >= 1; m >>= 1) rs += __shfl_xor(rs, m, 32);
        float corr = __expf(mst[r] - nm);
        lst[r] = lst[r] * corr + rs;
        mst[r] = nm;
#pragma unroll
        for (int nb = 0; nb < 4; nb++) acc[nb][r] *= corr;
        int Mrow = r + 8 * g;
        Ps[w * 512 + Mrow * 32 + lanen]      = (_Float16)p0;
        Ps[w * 512 + Mrow * 32 + 16 + lanen] = (_Float16)p1;
      }
      // ---- re-layout P (C-layout -> A-fragment) via per-wave LDS ----
      FragA pf;
      {
        const _Float16* pp = &Ps[w * 512 + lanen * 32];
        pf.h[0] = *(const v8h*)&pp[g * 8];
        pf.h[1] = *(const v8h*)&pp[16 + g * 8];
      }
      // ---- accumulate P(16x32) * V(32x64) ----
#pragma unroll
      for (int nb = 0; nb < 4; nb++) {
        FragA vf;
        const _Float16* vp = &Vt[(nb * 16 + lanen) * CHUNK];
        vf.h[0] = *(const v8h*)&vp[kt + g * 8];
        vf.h[1] = *(const v8h*)&vp[kt + 16 + g * 8];
        acc[nb] = wmma_f16(pf.v, vf.v, acc[nb]);
      }
    }
  }

  // ---- normalize and store (S x 512 f16, head-major columns) ----
#pragma unroll
  for (int r = 0; r < 8; r++) {
    float inv = 1.0f / lst[r];
    int row = seg * SEGL + qb + r + 8 * g;
#pragma unroll
    for (int nb = 0; nb < 4; nb++) {
      int col = head * DH + nb * 16 + lanen;
      ao[(size_t)row * (HEADS * DH) + col] = (_Float16)(acc[nb][r] * inv);
    }
  }
}

// ---------------------------------------------------------------------------
// Launch
// ---------------------------------------------------------------------------
extern "C" void kernel_launch(void* const* d_in, const int* in_sizes, int n_in,
                              void* d_out, int out_size, void* d_ws, size_t ws_size,
                              hipStream_t stream) {
  (void)in_sizes; (void)n_in; (void)out_size; (void)ws_size;
  const float* seq   = (const float*)d_in[0];
  const float* normw = (const float*)d_in[1];
  const float* wqkv  = (const float*)d_in[2];
  const float* wout  = (const float*)d_in[3];
  const float* pm    = (const float*)d_in[4];
  float* out = (float*)d_out;

  char* ws = (char*)d_ws;
  size_t off = 0;
  auto carve = [&](size_t bytes) -> char* {
    char* p = ws + off;
    off += (bytes + 255) & ~(size_t)255;
    return p;
  };
  _Float16* xh    = (_Float16*)carve((size_t)SQ * DIMM * 2);          // 4 MB
  _Float16* wqkvT = (_Float16*)carve((size_t)3 * DIMM * DIMM * 2);    // 1.5 MB
  _Float16* woutT = (_Float16*)carve((size_t)DIMM * DIMM * 2);        // 0.5 MB
  _Float16* qkvh  = (_Float16*)carve((size_t)SQ * 3 * DIMM * 2);      // 12 MB
  _Float16* qh    = (_Float16*)carve((size_t)HEADS * SQ * DH * 2);    // 4 MB
  _Float16* kh2   = (_Float16*)carve((size_t)HEADS * KLTOT * DH * 2); // 4 MB
  _Float16* vh2   = (_Float16*)carve((size_t)HEADS * KLTOT * DH * 2); // 4 MB
  _Float16* aoh   = (_Float16*)carve((size_t)SQ * DIMM * 2);          // 4 MB

  rmsnorm_kernel<<<SQ, 256, 0, stream>>>(seq, normw, xh);
  cvt_transpose_kernel<<<(DIMM * 3 * DIMM) / 256, 256, 0, stream>>>(wqkv, wqkvT, DIMM, 3 * DIMM);
  cvt_transpose_kernel<<<(DIMM * DIMM) / 256, 256, 0, stream>>>(wout, woutT, DIMM, DIMM);
  prep_pm_kernel<<<(HEADS * NPM * 32) / 256, 256, 0, stream>>>(pm, kh2, vh2);
  gemm_wmma_kernel<false><<<dim3((3 * DIMM) / 128, SQ / 128), 256, 0, stream>>>(
      xh, wqkvT, (void*)qkvh, SQ, 3 * DIMM, DIMM);
  rope_split_kernel<<<SQ, 256, 0, stream>>>(qkvh, qh, kh2, vh2);
  attn_kernel<<<dim3(4, 8, 8), 256, 0, stream>>>(qh, kh2, vh2, aoh);
  gemm_wmma_kernel<true><<<dim3(DIMM / 128, SQ / 128), 256, 0, stream>>>(
      aoh, woutT, (void*)out, SQ, DIMM, DIMM);
}
